// SparseLookupFFNv4_51934744543459
// MI455X (gfx1250) — compile-verified, matrix operand order checked by reading, not run on hardware
//
#include <hip/hip_runtime.h>
#include <hip/hip_bf16.h>
#include <math.h>

// ---------------------------------------------------------------------------
// SparseLookupFFN fused kernel for gfx1250 (MI455X, CDNA5, wave32).
//
// Shapes: B=4, T=2048, D=1024, H=256, NT=32, G=16, NS=2, SD=8. N = 8192 rows.
// HBM-bound (~32MB in + ~32MB out ~= 2.8us @ 23.3 TB/s; x re-read hits L2):
// one fused pass over x; the two GEMMs (xn@W1 [Kx256], xn@sign(dirs).T [Kx32])
// run on v_wmma_f32_16x16x32_f16 from LDS-staged f16 activations; the output
// stream is written through the CDNA5 async-LDS store engine (ASYNCcnt).
// NOTE: reference's softmax/tile_weights are dead code -> skipped.
// ---------------------------------------------------------------------------

typedef __attribute__((ext_vector_type(16))) _Float16 v16h;
typedef __attribute__((ext_vector_type(8)))  float    v8f;

// async-LDS builtin pointer payload types (per hipcc diagnostics)
typedef int ai2 __attribute__((vector_size(8)));    // b64 payload
typedef int ai4 __attribute__((vector_size(16)));   // b128 payload

#define D_DIM   1024
#define H_DIM   256
#define NT_T    32
#define G_Q     16
#define ROWS_PB 16            // rows per block
#define KTILES  (D_DIM / 32)  // 32 k-tiles of K=32
#define NTILES_H (H_DIM / 16) // 16 n-tiles for h
#define XN_STRIDE (D_DIM + 8) // half stride, 2064B: 16B aligned, bank-skewed
#define H_STRIDE  (H_DIM + 8)

#if defined(__gfx1250__) &&                                        \
    __has_builtin(__builtin_amdgcn_global_store_async_from_lds_b128) && \
    __has_builtin(__builtin_amdgcn_global_load_async_to_lds_b64)
#define HAS_ASYNC_LDS 1
#else
#define HAS_ASYNC_LDS 0
#endif

#if HAS_ASYNC_LDS
#if __has_builtin(__builtin_amdgcn_s_wait_asynccnt)
#define WAIT_ASYNC(n) __builtin_amdgcn_s_wait_asynccnt(n)
#else
#define WAIT_ASYNC(n) asm volatile("s_wait_asynccnt %0" :: "i"(n) : "memory")
#endif
#endif

union FragU {
    v16h v;
    uint4 q[2];
};

// K index inside a 32-wide k-tile for 16-bit WMMA A/B fragments:
// lane group g = lane>>4; half h in [0,16): k = (h&7) + (h>=8 ? 16 : 0) + 8*g
__device__ __forceinline__ int wmma_k(int h, int g) {
    return (h & 7) + ((h & 8) ? 16 : 0) + (g << 3);
}

__device__ __forceinline__ float bspline(float t) {
    t = fabsf(t);
    if (t < 1.0f) return 2.0f / 3.0f - t * t + 0.5f * t * t * t;
    if (t < 2.0f) { float u = 2.0f - t; return (1.0f / 6.0f) * u * u * u; }
    return 0.0f;
}

__device__ __forceinline__ float ternary1(float v) {
    return (v > 0.3f) ? 1.0f : ((v < -0.3f) ? -1.0f : 0.0f);
}

// --------------------------- prep kernels ----------------------------------

// Pack W1 (D x H, f32) into WMMA-B fragment order, f16.
__global__ void pack_w1_kernel(const float* __restrict__ W1,
                               _Float16* __restrict__ out) {
    int lane = threadIdx.x;
    int bid  = blockIdx.x;           // kt*16 + nt
    int kt   = bid >> 4;
    int g    = lane >> 4;
    int n    = (bid & 15) * 16 + (lane & 15);
    _Float16* o = out + ((size_t)bid * 32 + lane) * 16;
#pragma unroll
    for (int h = 0; h < 16; ++h) {
        int k = kt * 32 + wmma_k(h, g);
        o[h] = (_Float16)W1[(size_t)k * H_DIM + n];
    }
}

// Pack sign(directions).T (D x NT) into WMMA-B fragment order, f16.
__global__ void pack_sig_kernel(const float* __restrict__ directions,
                                _Float16* __restrict__ out) {
    int lane = threadIdx.x;
    int bid  = blockIdx.x;           // kt*2 + nt
    int kt   = bid >> 1;
    int g    = lane >> 4;
    int n    = (bid & 1) * 16 + (lane & 15);   // tile index
    _Float16* o = out + ((size_t)bid * 32 + lane) * 16;
#pragma unroll
    for (int h = 0; h < 16; ++h) {
        int k = kt * 32 + wmma_k(h, g);
        float d = directions[(size_t)n * D_DIM + k];
        o[h] = (_Float16)((d > 0.0f) ? 1.0f : ((d < 0.0f) ? -1.0f : 0.0f));
    }
}

// temporal[s][t] = sigmoid(state_emb[s] . state_signatures[t]); 1 block x 64
__global__ void temporal_kernel(const float* __restrict__ state_emb,
                                const float* __restrict__ state_sig,
                                float* __restrict__ out) {
    int tid = threadIdx.x;           // s*32 + t
    if (tid >= 2 * NT_T) return;
    int s = tid >> 5, t = tid & 31;
    float acc = 0.0f;
#pragma unroll
    for (int d = 0; d < 8; ++d) acc += state_emb[s * 8 + d] * state_sig[t * 8 + d];
    out[tid] = 1.0f / (1.0f + expf(-acc));
}

// --------------------------- main fused kernel -----------------------------

// LDS overlay: xn staging (stages 1-2) shares space with per-wave async
// output row buffers (stage 3).
union alignas(16) SharedU {
    _Float16 xn[ROWS_PB][XN_STRIDE];   // 33,024 B (stages 1-2)
    float    outbuf[4][2][D_DIM];      // 32,768 B (stage 3, dbl-buffered)
};

__global__ void __launch_bounds__(128)
sparse_lookup_ffn_kernel(const float* __restrict__ x,
                         const float* __restrict__ positions,
                         const int*   __restrict__ states,
                         const float* __restrict__ ln_gamma,
                         const float* __restrict__ ln_beta,
                         const float* __restrict__ b1,
                         const float* __restrict__ W2,
                         const float* __restrict__ b2,
                         const float* __restrict__ spline_coeffs,
                         const float* __restrict__ spline_scales,
                         const float* __restrict__ directions,
                         const float* __restrict__ state_mod,
                         const float* __restrict__ output_scale,
                         const _Float16* __restrict__ w1p,
                         const _Float16* __restrict__ sigp,
                         const float* __restrict__ temporal,
                         float* __restrict__ outp) {
    __shared__ SharedU  u;                            // 33,024 B
    __shared__ _Float16 lds_h [ROWS_PB][H_STRIDE];    //  8,448 B
    __shared__ float    lds_ct[ROWS_PB][NT_T];        //  2,048 B
    __shared__ float    lds_tm[2 * NT_T];             //    256 B

    const int lane = threadIdx.x & 31;
    const int wave = threadIdx.x >> 5;               // 4 waves
    const size_t row0 = (size_t)blockIdx.x * ROWS_PB;

    // ---- temporal gate table -> LDS (async engine on wave 0) ----
#if HAS_ASYNC_LDS
    if (wave == 0) {
        __builtin_amdgcn_global_load_async_to_lds_b64(
            (ai2*)(temporal + lane * 2), (ai2*)(lds_tm + lane * 2), 0, 0);
    }
#else
    if (threadIdx.x < 2 * NT_T) lds_tm[threadIdx.x] = temporal[threadIdx.x];
#endif

    // ---- Stage 1: LayerNorm 4 rows per wave, store xn as f16 to LDS ----
    for (int rr = 0; rr < 4; ++rr) {
        const int row = wave * 4 + rr;
        const size_t nrow = row0 + row;
        const float4* xr = (const float4*)(x + nrow * D_DIM);
        float4 vals[8];
        float s = 0.0f, s2 = 0.0f;
#pragma unroll
        for (int v = 0; v < 8; ++v) {
            float4 t = xr[lane + v * 32];
            vals[v] = t;
            s  += t.x + t.y + t.z + t.w;
            s2 += t.x * t.x + t.y * t.y + t.z * t.z + t.w * t.w;
        }
#pragma unroll
        for (int off = 16; off > 0; off >>= 1) {
            s  += __shfl_xor(s, off, 32);
            s2 += __shfl_xor(s2, off, 32);
        }
        const float mu  = s * (1.0f / D_DIM);
        const float var = s2 * (1.0f / D_DIM) - mu * mu;
        const float inv = rsqrtf(var + 1e-5f);
#pragma unroll
        for (int v = 0; v < 8; ++v) {
            const int base = (lane + v * 32) * 4;
            float4 t = vals[v];
            u.xn[row][base + 0] = (_Float16)((t.x - mu) * inv * ln_gamma[base + 0] + ln_beta[base + 0]);
            u.xn[row][base + 1] = (_Float16)((t.y - mu) * inv * ln_gamma[base + 1] + ln_beta[base + 1]);
            u.xn[row][base + 2] = (_Float16)((t.z - mu) * inv * ln_gamma[base + 2] + ln_beta[base + 2]);
            u.xn[row][base + 3] = (_Float16)((t.w - mu) * inv * ln_gamma[base + 3] + ln_beta[base + 3]);
        }
    }
#if HAS_ASYNC_LDS
    if (wave == 0) WAIT_ASYNC(0);     // temporal table landed in LDS
#endif
    __syncthreads();

    // ---- Stage 2: WMMA GEMMs. Wave w owns h-ntiles {w, w+4, w+8, w+12};
    //      waves 0/1 additionally own content ntile w (NT=32 = two tiles). ----
    const int row16 = lane & 15;
    const int g16   = lane >> 4;
    const v8f zero = {0.f, 0.f, 0.f, 0.f, 0.f, 0.f, 0.f, 0.f};
    v8f acc0 = zero, acc1 = zero, acc2 = zero, acc3 = zero, accC = zero;

    for (int kt = 0; kt < KTILES; ++kt) {
        // A fragment: 16x32 f16 tile from LDS (ISA layout: two 16B reads)
        FragU a;
        const _Float16* arow = &u.xn[row16][kt * 32 + g16 * 8];
        a.q[0] = *reinterpret_cast<const uint4*>(arow);
        a.q[1] = *reinterpret_cast<const uint4*>(arow + 16);

        if (kt + 1 < KTILES) {
            __builtin_prefetch(w1p + (((size_t)(kt + 1) * 16 + wave) * 32 + lane) * 16, 0, 1);
        }
#pragma unroll
        for (int i = 0; i < 4; ++i) {
            const int nt = wave + i * 4;
            FragU b;
            const uint4* bp = reinterpret_cast<const uint4*>(
                w1p + (((size_t)kt * 16 + nt) * 32 + lane) * 16);
            b.q[0] = bp[0];
            b.q[1] = bp[1];
            v8f* accp = (i == 0) ? &acc0 : (i == 1) ? &acc1 : (i == 2) ? &acc2 : &acc3;
            *accp = __builtin_amdgcn_wmma_f32_16x16x32_f16(
                false, a.v, false, b.v, (short)0, *accp, false, false);
        }
        if (wave < 2) {
            FragU sg;
            const uint4* sp = reinterpret_cast<const uint4*>(
                sigp + (((size_t)kt * 2 + wave) * 32 + lane) * 16);
            sg.q[0] = sp[0];
            sg.q[1] = sp[1];
            accC = __builtin_amdgcn_wmma_f32_16x16x32_f16(
                false, a.v, false, sg.v, (short)0, accC, false, false);
        }
    }

    // h epilogue: bias + exact gelu, store f16 to LDS.
    // C/D layout: VGPR r, lanes0-15 -> M=r, lanes16-31 -> M=r+8; N = lane&15.
#pragma unroll
    for (int i = 0; i < 4; ++i) {
        const int nt = wave + i * 4;
        const int n  = nt * 16 + row16;
        const float bias = b1[n];
        const v8f acc = (i == 0) ? acc0 : (i == 1) ? acc1 : (i == 2) ? acc2 : acc3;
#pragma unroll
        for (int r = 0; r < 8; ++r) {
            const int m = r + 8 * g16;
            const float v = acc[r] + bias;
            const float ge = 0.5f * v * (1.0f + erff(v * 0.70710678f));
            lds_h[m][n] = (_Float16)ge;
        }
    }
    if (wave < 2) {
        const int n = wave * 16 + row16;
#pragma unroll
        for (int r = 0; r < 8; ++r) lds_ct[r + 8 * g16][n] = accC[r];
    }
    __syncthreads();   // xn dead from here on: outbuf overlay becomes live

    // ---- Stage 3: per-row epilogue. One lane per tile (NT==32==wave32). ----
    const float oscale = output_scale[0];
    for (int rr = 0; rr < 4; ++rr) {
        const int row = wave * 4 + rr;
        const size_t nrow = row0 + row;
        const float pos = positions[nrow];
        const int   st  = states[nrow];

        // combined = content * spatial * temporal, argmax over 32 tiles
        const float pn = pos * ((float)NT_T / 2048.0f);
        const float spatial = bspline((pn - (float)lane) * 0.5f);
        float bv = lds_ct[row][lane] * spatial * lds_tm[st * NT_T + lane];
        int bi = lane;
#pragma unroll
        for (int off = 16; off > 0; off >>= 1) {
            float ov = __shfl_xor(bv, off, 32);
            int   oi = __shfl_xor(bi, off, 32);
            if (ov > bv || (ov == bv && oi < bi)) { bv = ov; bi = oi; }
        }
        const int tile = bi;

        // comp = tanh(h @ W2 + b2): 256-wide dot, 8 elems/lane
        float pa = 0.0f, pb = 0.0f;
#pragma unroll
        for (int kk = 0; kk < 8; ++kk) {
            const int k = lane + kk * 32;
            const float hv = (float)lds_h[row][k];
            pa += hv * W2[k * 2 + 0];
            pb += hv * W2[k * 2 + 1];
        }
#pragma unroll
        for (int off = 16; off > 0; off >>= 1) {
            pa += __shfl_xor(pa, off, 32);
            pb += __shfl_xor(pb, off, 32);
        }
        const float aC = tanhf(pa + b2[0]);
        const float bC = tanhf(pb + b2[1]);
        int ia = (int)((aC + 1.0f) * 8.0f);  ia = ia < 0 ? 0 : (ia > 15 ? 15 : ia);
        int ib = (int)((bC + 1.0f) * 8.0f);  ib = ib < 0 ? 0 : (ib > 15 ? 15 : ib);
        const float cs = 2.0f / (float)G_Q;
        const float la = (aC + 1.0f - (float)ia * cs) * (1.0f / cs);
        const float lb = (bC + 1.0f - (float)ib * cs) * (1.0f / cs);
        const float* cc = spline_coeffs + (((size_t)tile * G_Q + ia) * G_Q + ib) * 3;
        const float scale = (ternary1(cc[0]) + ternary1(cc[1]) * la + ternary1(cc[2]) * lb)
                            * spline_scales[tile];
        const float smod = state_mod[st * NT_T + tile];
        const float sfin = scale * smod * oscale;

        // out = x + sfin * directions[tile]  (float4 stream, 8 per lane)
        const float4* xr = (const float4*)(x + nrow * D_DIM);
        const float4* dr = (const float4*)(directions + (size_t)tile * D_DIM);
#if HAS_ASYNC_LDS
        // Build the row in LDS, then fire-and-forget via the async engine.
        float* lb0 = &u.outbuf[wave][rr & 1][0];
        float4* lb4 = (float4*)lb0;
        WAIT_ASYNC(8);   // allow prev row in flight; row rr-2's buffer is free
#pragma unroll
        for (int v = 0; v < 8; ++v) {
            const int i4 = lane + v * 32;
            float4 xv = xr[i4];
            float4 dv = dr[i4];
            float4 ov;
            ov.x = xv.x + sfin * dv.x;
            ov.y = xv.y + sfin * dv.y;
            ov.z = xv.z + sfin * dv.z;
            ov.w = xv.w + sfin * dv.w;
            lb4[i4] = ov;
        }
        asm volatile("s_wait_dscnt 0x0" ::: "memory");  // LDS writes visible
        float* gb = outp + nrow * D_DIM;
#pragma unroll
        for (int v = 0; v < 8; ++v) {
            // one issue = 16B per lane = 512B per wave
            __builtin_amdgcn_global_store_async_from_lds_b128(
                (ai4*)(gb + lane * 4 + v * 128),
                (ai4*)(lb0 + lane * 4 + v * 128), 0, 0);
        }
#else
        float4* orow = (float4*)(outp + nrow * D_DIM);
#pragma unroll
        for (int v = 0; v < 8; ++v) {
            const int i4 = lane + v * 32;
            float4 xv = xr[i4];
            float4 dv = dr[i4];
            float4 ov;
            ov.x = xv.x + sfin * dv.x;
            ov.y = xv.y + sfin * dv.y;
            ov.z = xv.z + sfin * dv.z;
            ov.w = xv.w + sfin * dv.w;
            orow[i4] = ov;
        }
#endif
    }
    // S_ENDPGM performs an implicit wait-idle (all ASYNCcnt drained).
}

// --------------------------- launcher --------------------------------------

extern "C" void kernel_launch(void* const* d_in, const int* in_sizes, int n_in,
                              void* d_out, int out_size, void* d_ws, size_t ws_size,
                              hipStream_t stream) {
    (void)in_sizes; (void)n_in; (void)out_size; (void)ws_size;
    const float* x              = (const float*)d_in[0];
    const float* positions      = (const float*)d_in[1];
    const int*   states         = (const int*)  d_in[2];
    const float* ln_gamma       = (const float*)d_in[3];
    const float* ln_beta        = (const float*)d_in[4];
    const float* W1             = (const float*)d_in[5];
    const float* b1             = (const float*)d_in[6];
    const float* W2             = (const float*)d_in[7];
    const float* b2             = (const float*)d_in[8];
    const float* spline_coeffs  = (const float*)d_in[9];
    const float* spline_scales  = (const float*)d_in[10];
    const float* directions     = (const float*)d_in[11];
    const float* state_sigs     = (const float*)d_in[12];
    const float* state_emb      = (const float*)d_in[13];
    const float* state_mod      = (const float*)d_in[14];
    const float* output_scale   = (const float*)d_in[15];
    float* outp = (float*)d_out;

    // workspace layout
    _Float16* w1p   = (_Float16*)d_ws;                                   // 512 KB
    _Float16* sigp  = (_Float16*)((char*)d_ws + (512u << 10));           //  64 KB
    float*    tempo = (float*)   ((char*)d_ws + (512u << 10) + (64u << 10)); // 256 B

    pack_w1_kernel <<<KTILES * NTILES_H, 32, 0, stream>>>(W1, w1p);
    pack_sig_kernel<<<KTILES * 2,        32, 0, stream>>>(directions, sigp);
    temporal_kernel<<<1, 64, 0, stream>>>(state_emb, state_sigs, tempo);

    const int nrows = 4 * 2048;  // B*T
    sparse_lookup_ffn_kernel<<<nrows / ROWS_PB, 128, 0, stream>>>(
        x, positions, states, ln_gamma, ln_beta, b1, W2, b2,
        spline_coeffs, spline_scales, directions, state_mod, output_scale,
        w1p, sigp, tempo, outp);
}
